// SCD_cell_14061722927472
// MI455X (gfx1250) — compile-verified, matrix operand order checked by reading, not run on hardware
//
#include <hip/hip_runtime.h>
#include <hip/hip_bf16.h>

// ---------------------------------------------------------------------------
// Problem constants
// ---------------------------------------------------------------------------
#define H     1024
#define B     4096
#define KTOT  3072           // 3 * H  (fused K dimension)
#define ROWS  4097           // 4*H + 1
#define KC    128            // K-chunk staged in LDS per stage
#define NSTAGE (KTOT / KC)   // 24

// workspace layout: [ Acat bf16 (4097*3072) ][ Bcat bf16 (3072*4096, col-major) ]
#define ACAT_ELEMS ((size_t)ROWS * KTOT)
#define ACAT_BYTES (ACAT_ELEMS * 2)

typedef __attribute__((ext_vector_type(16))) __bf16 v16bf;
typedef __attribute__((ext_vector_type(8)))  __bf16 bf16x8;
typedef __attribute__((ext_vector_type(8)))  float  v8f;
typedef int v4i_vs __attribute__((__vector_size__(16)));   // matches builtin param

union AFrag { v16bf v; bf16x8 h[2]; };
struct BPair { AFrag b0, b1; };

#define WMMA(A, Bf, C) \
    __builtin_amdgcn_wmma_f32_16x16x32_bf16(false, (A), false, (Bf), (short)0, (C), false, false)

// gfx1250 async global->LDS copy path (probe with __has_builtin; reg fallback)
#if defined(__has_builtin)
#  if __has_builtin(__builtin_amdgcn_global_load_async_to_lds_b128) && \
      __has_builtin(__builtin_amdgcn_s_wait_asynccnt)
#    define USE_ASYNC_LDS 1
#  endif
#endif
#ifndef USE_ASYNC_LDS
#  define USE_ASYNC_LDS 0
#endif
#define AS3 __attribute__((address_space(3)))

__device__ __forceinline__ void lds_frag(AFrag& a, const __bf16* p) {
    a.h[0] = *(const bf16x8*)(p);
    a.h[1] = *(const bf16x8*)(p + 16);
}

__device__ __forceinline__ void load_B(BPair& bp, const __bf16* pB0,
                                       const __bf16* pB1, int kk) {
    bp.b0.h[0] = *(const bf16x8*)(pB0 + kk);
    bp.b0.h[1] = *(const bf16x8*)(pB0 + kk + 16);
    bp.b1.h[0] = *(const bf16x8*)(pB1 + kk);
    bp.b1.h[1] = *(const bf16x8*)(pB1 + kk + 16);
}

// ---------------------------------------------------------------------------
// Kernel 1: pack weights [U11 | U21 | W01] -> bf16 Acat, row-major, stride KTOT
// ---------------------------------------------------------------------------
__global__ void __launch_bounds__(256)
pack_a(const float* __restrict__ U11, const float* __restrict__ U21,
       const float* __restrict__ W01, __bf16* __restrict__ Acat) {
    size_t e = ((size_t)blockIdx.x * 256 + threadIdx.x) * 4;   // 4 elems / thread
    if (e >= ACAT_ELEMS) return;
    int r  = (int)(e / KTOT);
    int kk = (int)(e % KTOT);
    const float* src = (kk < H) ? U11 : (kk < 2 * H ? U21 : W01);
    int kloc = kk & (H - 1);
    float4 v = *(const float4*)(src + (size_t)r * H + kloc);
    Acat[e + 0] = (__bf16)v.x;
    Acat[e + 1] = (__bf16)v.y;
    Acat[e + 2] = (__bf16)v.z;
    Acat[e + 3] = (__bf16)v.w;
}

// ---------------------------------------------------------------------------
// Kernel 2: gate-prescale activations, bf16-convert, transpose via LDS.
// Bcat is column-major: element (k, n) at n*KTOT + k  -> contiguous K per col.
// ---------------------------------------------------------------------------
__global__ void __launch_bounds__(256)
pack_b(const float* __restrict__ h,    const float* __restrict__ htop,
       const float* __restrict__ hbot, const float* __restrict__ z,
       const float* __restrict__ zb,   __bf16* __restrict__ Bcat) {
    __shared__ __bf16 tile[32][33];
    int kt0 = (blockIdx.x % (KTOT / 32)) * 32;   // 96 k-tiles
    int nt0 = (blockIdx.x / (KTOT / 32)) * 32;   // 128 n-tiles
    int tx = threadIdx.x & 31;
    int ty = threadIdx.x >> 5;                   // 0..7
    int n  = nt0 + tx;
    float zc  = z[n];
    float zbc = zb[n];
    #pragma unroll
    for (int r = 0; r < 4; ++r) {
        int k = kt0 + ty + r * 8;
        float val;
        if (k < H)            val = h   [(size_t)k          * B + n] * (1.0f - zc);
        else if (k < 2 * H)   val = htop[(size_t)(k - H)    * B + n] * zc;
        else                  val = hbot[(size_t)(k - 2*H)  * B + n] * zbc;
        tile[ty + r * 8][tx] = (__bf16)val;
    }
    __syncthreads();
    #pragma unroll
    for (int r = 0; r < 4; ++r) {
        int col = nt0 + ty + r * 8;
        int k   = kt0 + tx;
        Bcat[(size_t)col * KTOT + k] = tile[tx][ty + r * 8];
    }
}

// ---------------------------------------------------------------------------
// Fused epilogue for one 16x16 tile (all four gates)
// ---------------------------------------------------------------------------
__device__ __forceinline__ void epilogue(
    const v8f& af, const v8f& ai, const v8f& ao, const v8f& ag,
    int m0, int rhalf, int n,
    const float* __restrict__ bias, const float* __restrict__ c,
    const float* __restrict__ hprev, float zc, float zbc,
    float* __restrict__ out) {
    const float omz  = 1.0f - zc;
    const float omzb = 1.0f - zbc;
    #pragma unroll
    for (int v = 0; v < 8; ++v) {
        int m = m0 + v + rhalf;
        float fs_f = af[v] + bias[m];
        float fs_i = ai[v] + bias[m + H];
        float fs_o = ao[v] + bias[m + 2 * H];
        float fs_g = ag[v] + bias[m + 3 * H];

        float fg = 1.0f / (1.0f + __expf(-fs_f));
        float ig = 1.0f / (1.0f + __expf(-fs_i));
        float og = 1.0f / (1.0f + __expf(-fs_o));
        float gg = tanhf(fs_g);

        size_t idx = (size_t)m * B + n;
        float cold = c[idx];
        float hold = hprev[idx];

        float iga  = ig * gg;
        float cnew = zc * iga + omz * omzb * cold + omz * zbc * (fg * cold + iga);
        float tc   = tanhf(cnew);
        float hnew = zc * og * tc + omz * omzb * hold + omz * zbc * og * tc;

        out[idx] = hnew;                       // h_new
        out[(size_t)H * B + idx] = cnew;       // c_new
    }
}

// ---------------------------------------------------------------------------
// Kernel 3: fused GEMM + LSTM epilogue, LDS-staged A.
// Block = 8 waves sharing one 16-row m-tile (64 A rows incl. 4 gates) over
// 256 columns (32 per wave).  A is staged into double-buffered LDS in 128-K
// chunks via async global->LDS copies; B streams from L2 one step ahead.
// ---------------------------------------------------------------------------
__global__ void __launch_bounds__(256)
gemm_fused(const __bf16* __restrict__ Acat, const __bf16* __restrict__ Bcat,
           const float* __restrict__ bias,  const float* __restrict__ z,
           const float* __restrict__ zbot,  const float* __restrict__ c,
           const float* __restrict__ hprev, float* __restrict__ out) {
    __shared__ __bf16 Abuf[2][64][KC + 8];    // +8 pad: even bank spread

    const int tid  = threadIdx.x;
    const int lane = tid & 31;
    const int wave = tid >> 5;
    const int mt = blockIdx.x >> 4;           // 0..63  m-tile
    const int cg = blockIdx.x & 15;           // 0..15  256-col group
    const int m0 = mt * 16;
    const int n0 = cg * 256 + wave * 32;

    const int l    = lane & 15;
    const int koff = (lane >= 16) ? 8 : 0;    // per-ISA 16-bit A/B fragment layout

    // ---- A staging mapping: thread -> (row 0..63, 16B segment) ------------
    const int arow   = tid >> 2;                               // 0..63
    const int absrow = (arow >> 4) * H + m0 + (arow & 15);     // gate*H + m-row
    const __bf16* asrc = Acat + (size_t)absrow * KTOT + (tid & 3) * 8;
    __bf16* adst = &Abuf[0][arow][(tid & 3) * 8];
    const size_t abufstride = (size_t)64 * (KC + 8);           // elems per buffer

    // ---- B pointers -------------------------------------------------------
    const __bf16* pB0 = Bcat + ((size_t)(n0 + l))      * KTOT + koff;
    const __bf16* pB1 = Bcat + ((size_t)(n0 + 16 + l)) * KTOT + koff;

    // ---- stage 0 ----------------------------------------------------------
#if USE_ASYNC_LDS
    #pragma unroll
    for (int j = 0; j < 4; ++j)
        __builtin_amdgcn_global_load_async_to_lds_b128(
            (v4i_vs*)(asrc + j * 32), (AS3 v4i_vs*)(adst + j * 32), 0, 0);
#else
    {
        bf16x8 stg[4];
        #pragma unroll
        for (int j = 0; j < 4; ++j) stg[j] = *(const bf16x8*)(asrc + j * 32);
        #pragma unroll
        for (int j = 0; j < 4; ++j) *(bf16x8*)(adst + j * 32) = stg[j];
    }
#endif

    BPair bb0, bb1;
    load_B(bb0, pB0, pB1, 0);

    v8f f0 = {}, f1 = {}, i0 = {}, i1 = {};
    v8f o0 = {}, o1 = {}, g0 = {}, g1 = {};

#if USE_ASYNC_LDS
    __builtin_amdgcn_s_wait_asynccnt(0);
#endif
    __syncthreads();

    for (int s = 0; s < NSTAGE; ++s) {
        const int buf  = s & 1;
        const int kc   = s * KC;
        const int snxt = (s + 1 < NSTAGE) ? (s + 1) : s;   // clamp: harmless redo
        const int kcn  = snxt * KC;

        // issue next stage's A copy (writes the other LDS buffer)
        __bf16* adstn = adst + ((snxt & 1) ? abufstride : 0);
#if USE_ASYNC_LDS
        #pragma unroll
        for (int j = 0; j < 4; ++j)
            __builtin_amdgcn_global_load_async_to_lds_b128(
                (v4i_vs*)(asrc + kcn + j * 32), (AS3 v4i_vs*)(adstn + j * 32), 0, 0);
#else
        bf16x8 stg[4];
        #pragma unroll
        for (int j = 0; j < 4; ++j) stg[j] = *(const bf16x8*)(asrc + kcn + j * 32);
#endif
        __builtin_prefetch(pB0 + kc + 2 * KC, 0, 1);

        // 4 k-steps of 32 on the current LDS buffer
        #pragma unroll
        for (int js = 0; js < 4; ++js) {
            const int kk = kc + js * 32;
            int knext = kk + 32;
            if (knext >= KTOT) knext = 0;            // branchless wrap (last step)
            if (js & 1) load_B(bb0, pB0, pB1, knext);
            else        load_B(bb1, pB0, pB1, knext);

            AFrag a0, a1, a2, a3;
            const int ks = js * 32 + koff;
            lds_frag(a0, &Abuf[buf][ 0 + l][ks]);
            lds_frag(a1, &Abuf[buf][16 + l][ks]);
            lds_frag(a2, &Abuf[buf][32 + l][ks]);
            lds_frag(a3, &Abuf[buf][48 + l][ks]);

            const BPair& cur = (js & 1) ? bb1 : bb0;
            f0 = WMMA(a0.v, cur.b0.v, f0);
            i0 = WMMA(a1.v, cur.b0.v, i0);
            o0 = WMMA(a2.v, cur.b0.v, o0);
            g0 = WMMA(a3.v, cur.b0.v, g0);
            f1 = WMMA(a0.v, cur.b1.v, f1);
            i1 = WMMA(a1.v, cur.b1.v, i1);
            o1 = WMMA(a2.v, cur.b1.v, o1);
            g1 = WMMA(a3.v, cur.b1.v, g1);
        }

#if USE_ASYNC_LDS
        __builtin_amdgcn_s_wait_asynccnt(0);
#else
        #pragma unroll
        for (int j = 0; j < 4; ++j) *(bf16x8*)(adstn + j * 32) = stg[j];
#endif
        __syncthreads();
    }

    // ---- fused epilogue -------------------------------------------------
    const int rhalf = (lane >= 16) ? 8 : 0;
    const int ncol0 = n0 + (lane & 15);
    const int ncol1 = n0 + 16 + (lane & 15);
    epilogue(f0, i0, o0, g0, m0, rhalf, ncol0, bias, c, hprev,
             z[ncol0], zbot[ncol0], out);
    epilogue(f1, i1, o1, g1, m0, rhalf, ncol1, bias, c, hprev,
             z[ncol1], zbot[ncol1], out);
}

// ---------------------------------------------------------------------------
// Kernel 4: z_hat row (row 4096) — fp32 dot products, one wave per column.
// ---------------------------------------------------------------------------
__global__ void __launch_bounds__(256)
zhat_row(const float* __restrict__ U11, const float* __restrict__ U21,
         const float* __restrict__ W01, const float* __restrict__ h,
         const float* __restrict__ htop, const float* __restrict__ hbot,
         const float* __restrict__ z,   const float* __restrict__ zb,
         const float* __restrict__ bias, float* __restrict__ out) {
    const int wid  = (blockIdx.x * 256 + threadIdx.x) >> 5;
    const int lane = threadIdx.x & 31;
    if (wid >= B) return;
    const int col = wid;
    const float s1 = 1.0f - z[col];
    const float s2 = z[col];
    const float s3 = zb[col];
    const float* u11 = U11 + (size_t)(ROWS - 1) * H;
    const float* u21 = U21 + (size_t)(ROWS - 1) * H;
    const float* w01 = W01 + (size_t)(ROWS - 1) * H;

    float acc = 0.0f;
    for (int k = lane; k < H; k += 32) {
        acc += s1 * u11[k] * h   [(size_t)k * B + col];
        acc += s2 * u21[k] * htop[(size_t)k * B + col];
        acc += s3 * w01[k] * hbot[(size_t)k * B + col];
    }
    #pragma unroll
    for (int off = 16; off > 0; off >>= 1)
        acc += __shfl_down(acc, off, 32);
    if (lane == 0) {
        float s  = acc + bias[ROWS - 1];
        float zh = fminf(fmaxf((s + 1.0f) * 0.5f, 0.0f), 1.0f);   // hard_sigm, ALPHA=1
        out[(size_t)2 * H * B + col] = zh;
    }
}

// ---------------------------------------------------------------------------
// Launch
// ---------------------------------------------------------------------------
extern "C" void kernel_launch(void* const* d_in, const int* in_sizes, int n_in,
                              void* d_out, int out_size, void* d_ws, size_t ws_size,
                              hipStream_t stream) {
    const float* c_in  = (const float*)d_in[0];
    const float* hbot  = (const float*)d_in[1];
    const float* h_in  = (const float*)d_in[2];
    const float* htop  = (const float*)d_in[3];
    const float* z     = (const float*)d_in[4];
    const float* zb    = (const float*)d_in[5];
    const float* U11   = (const float*)d_in[6];
    const float* U21   = (const float*)d_in[7];
    const float* W01   = (const float*)d_in[8];
    const float* bias  = (const float*)d_in[9];
    float* out = (float*)d_out;

    __bf16* Acat = (__bf16*)d_ws;
    __bf16* Bcat = (__bf16*)((char*)d_ws + ACAT_BYTES);

    // 1) weights -> bf16 (4 elems/thread)
    {
        int blocks = (int)((ACAT_ELEMS / 4 + 255) / 256);
        pack_a<<<blocks, 256, 0, stream>>>(U11, U21, W01, Acat);
    }
    // 2) activations -> gate-prescaled bf16, column-major
    {
        int blocks = (KTOT / 32) * (B / 32);   // 96 * 128
        pack_b<<<blocks, 256, 0, stream>>>(h_in, htop, hbot, z, zb, Bcat);
    }
    // 3) fused GEMM + LSTM cell update: 64 m-tiles * 16 col-groups
    {
        int blocks = 64 * 16;
        gemm_fused<<<blocks, 256, 0, stream>>>(Acat, Bcat, bias, z, zb,
                                               c_in, h_in, out);
    }
    // 4) z_hat row: 4096 waves
    {
        int blocks = (B * 32) / 256;
        zhat_row<<<blocks, 256, 0, stream>>>(U11, U21, W01, h_in, htop, hbot,
                                             z, zb, bias, out);
    }
}
